// IC_Decoder_angle_54142357733963
// MI455X (gfx1250) — compile-verified, matrix operand order checked by reading, not run on hardware
//
#include <hip/hip_runtime.h>

#define NN 20000
#define EE 640000
#define FDIM 128
#define SCF_REAL 138
#define SCF_PAD 160
#define PI_F 3.14159265358979323846f

typedef __attribute__((ext_vector_type(16))) __bf16       v16bf;
typedef __attribute__((ext_vector_type(8)))  float        v8f;
typedef __attribute__((ext_vector_type(4)))  unsigned int u32x4;

union BFrag {
  v16bf v;
  unsigned short h[16];
  u32x4 q[2];
};

// silu via fast hardware reciprocal (v_rcp_f32) instead of IEEE divide expansion
__device__ __forceinline__ float silu_f(float x) {
  return x * __builtin_amdgcn_rcpf(1.0f + __expf(-x));
}

__device__ __forceinline__ unsigned short f2bf(float f) {
  union { float f; unsigned int u; } cv; cv.f = f;
  unsigned int u = cv.u;
  u += 0x7fffu + ((u >> 16) & 1u);   // round-to-nearest-even
  return (unsigned short)(u >> 16);
}

// ---------------------------------------------------------------------------
// Weight pre-swizzle: f32 [L][K0][N0] row-major -> bf16 B-fragments.
// Fragment layout (wave32, v_wmma_f32_16x16x32_bf16 B operand, 32xK16 tile):
//   lane gives n = nt*16 + (lane&15); k-group base kb = (lane>=16)?8:0
//   element e (0..15): k = kt*32 + kb + (e<8 ? e : 16+(e-8))
// Stored per (l,kt,nt,lane) as 16 contiguous bf16 (32B) -> 2x b128 loads.
// ---------------------------------------------------------------------------
__global__ __launch_bounds__(256) void wconv_kernel(
    const float* __restrict__ W, unsigned short* __restrict__ Wf,
    int L, int K0, int N0, int KT, int NT)
{
  const int idx = blockIdx.x * 256 + threadIdx.x;
  const int total = L * KT * NT * 32;
  if (idx >= total) return;
  const int lane = idx & 31;
  int t = idx >> 5;
  const int nt = t % NT; t /= NT;
  const int kt = t % KT; t /= KT;
  const int l = t;
  const int kb = (lane >> 4) << 3;
  const int n = (nt << 4) + (lane & 15);
  unsigned short* o = Wf + (size_t)idx * 16;
#pragma unroll
  for (int e = 0; e < 16; ++e) {
    const int k = kt * 32 + kb + ((e < 8) ? e : (16 + (e - 8)));
    const float v = (k < K0 && n < N0) ? W[((size_t)l * K0 + k) * N0 + n] : 0.0f;
    o[e] = f2bf(v);
  }
}

__global__ __launch_bounds__(256) void zero_kernel(float* __restrict__ p, int n) {
  const int i = blockIdx.x * 256 + threadIdx.x;
  if (i < n) p[i] = 0.0f;
}

// S0 = concat(S[N,124], res_embed[cg_z][N,4]) -> [N,128]
__global__ __launch_bounds__(256) void embed_kernel(
    const float* __restrict__ S, const int* __restrict__ cgz,
    const float* __restrict__ res_embed, float* __restrict__ Sbuf)
{
  const int idx = blockIdx.x * 256 + threadIdx.x;
  if (idx >= NN * FDIM) return;
  const int n = idx >> 7, c = idx & 127;
  Sbuf[idx] = (c < 124) ? S[(size_t)n * 124 + c]
                        : res_embed[(size_t)cgz[n] * 4 + (c - 124)];
}

// ---------------------------------------------------------------------------
// Fused 2-layer MLP via WMMA bf16:  Y = [Res +] (act(act?(X)@W1+b1))@W2+b2
// 8 waves/block, each wave: 16 rows x all columns. Intermediate tile staged
// through LDS (bf16) to re-fragment for layer 2.
// ---------------------------------------------------------------------------
template<int KT1, int NT1, int KT2, int NT2, int XS, int OS, int NB1, int NB2,
         bool SILU_IN, bool RESID>
__global__ __launch_bounds__(256) void mlp2_kernel(
    const float* __restrict__ X, const float* __restrict__ Res, float* __restrict__ Y,
    const unsigned short* __restrict__ W1f, const float* __restrict__ b1,
    const unsigned short* __restrict__ W2f, const float* __restrict__ b2,
    int nrows)
{
  constexpr int MIDS = KT2 * 32;               // LDS row stride (halves), K pad for layer 2
  __shared__ unsigned short mid[8 * 16 * MIDS];
  const int lane = threadIdx.x & 31;
  const int wave = threadIdx.x >> 5;
  const int tile = blockIdx.x * 8 + wave;
  const int ntiles = nrows >> 4;
  const bool active = tile < ntiles;
  unsigned short* my = &mid[wave * 16 * MIDS];
  const int kb = (lane >> 4) << 3;             // 0 or 8
  const int nl = lane & 15;
  const int row0 = (lane >> 4) << 3;           // C/D: M = r + row0

  BFrag a1[KT1];
  if (active) {
    for (int i = lane; i < 16 * MIDS; i += 32) my[i] = 0;  // also zeroes K-pad cols
    const int m = (tile << 4) + nl;
    const float* xr = X + (size_t)m * XS;
#pragma unroll
    for (int t = 0; t < KT1; ++t) {
      const float* p = xr + t * 32 + kb;
#pragma unroll
      for (int e = 0; e < 8; ++e) {
        float v0 = p[e];
        float v1 = p[16 + e];
        if (SILU_IN) { v0 = silu_f(v0); v1 = silu_f(v1); }
        a1[t].h[e] = f2bf(v0);
        a1[t].h[8 + e] = f2bf(v1);
      }
    }
#pragma unroll 1
    for (int nt = 0; nt < NT1; ++nt) {
      const int n = (nt << 4) + nl;
      const float bv = (n < NB1) ? b1[n] : 0.0f;
      v8f acc = { bv, bv, bv, bv, bv, bv, bv, bv };
#pragma unroll
      for (int t = 0; t < KT1; ++t) {
        BFrag bf;
        const u32x4* wp = (const u32x4*)(W1f + (((size_t)(t * NT1 + nt) * 32) + lane) * 16);
        bf.q[0] = wp[0];
        bf.q[1] = wp[1];
        acc = __builtin_amdgcn_wmma_f32_16x16x32_bf16(
            false, a1[t].v, false, bf.v, (short)0, acc, false, false);
      }
#pragma unroll
      for (int r = 0; r < 8; ++r)
        my[(row0 + r) * MIDS + n] = f2bf(silu_f(acc[r]));
    }
  }
  __syncthreads();
  if (active) {
    BFrag a2[KT2];
#pragma unroll
    for (int t = 0; t < KT2; ++t) {
      const unsigned short* p = &my[nl * MIDS + t * 32 + kb];
      a2[t].q[0] = *(const u32x4*)p;
      a2[t].q[1] = *(const u32x4*)(p + 16);
    }
#pragma unroll 1
    for (int nt = 0; nt < NT2; ++nt) {
      const int n = (nt << 4) + nl;
      const float bv = (n < NB2) ? b2[n] : 0.0f;
      v8f acc = { bv, bv, bv, bv, bv, bv, bv, bv };
#pragma unroll
      for (int t = 0; t < KT2; ++t) {
        BFrag bf;
        const u32x4* wp = (const u32x4*)(W2f + (((size_t)(t * NT2 + nt) * 32) + lane) * 16);
        bf.q[0] = wp[0];
        bf.q[1] = wp[1];
        acc = __builtin_amdgcn_wmma_f32_16x16x32_bf16(
            false, a2[t].v, false, bf.v, (short)0, acc, false, false);
      }
#pragma unroll
      for (int r = 0; r < 8; ++r) {
        const size_t o = (size_t)((tile << 4) + row0 + r) * OS + n;
        float h = acc[r];
        if (RESID) h += Res[o];
        Y[o] = h;
      }
    }
  }
}

// ---------------------------------------------------------------------------
// Fused edge phase: geometry -> rbf/env -> w_s = (rbf@dist_W+b)*env ->
// gather phi[j] -> atomic scatter-add into v[i].  One wave per edge.
// ---------------------------------------------------------------------------
__global__ __launch_bounds__(256) void edge_kernel(
    const int* __restrict__ nbr, const float* __restrict__ xyz,
    const float* __restrict__ phi, float* __restrict__ vout,
    const float* __restrict__ distW, const float* __restrict__ distb)
{
  __shared__ float sW[16 * FDIM];
  __shared__ float sb[FDIM];
  for (int i = threadIdx.x; i < 16 * FDIM; i += 256) sW[i] = distW[i];
  for (int i = threadIdx.x; i < FDIM; i += 256) sb[i] = distb[i];
  __syncthreads();
  const int lane = threadIdx.x & 31;
  const int c0 = lane << 2;                    // 4 columns per lane
  const int w = blockIdx.x * 8 + (threadIdx.x >> 5);
  const int nw = gridDim.x * 8;
  for (int e = w; e < EE; e += nw) {
    const int ii = nbr[2 * e];
    const int jj = nbr[2 * e + 1];
    const float dx = xyz[3 * jj + 0] - xyz[3 * ii + 0];
    const float dy = xyz[3 * jj + 1] - xyz[3 * ii + 1];
    const float dz = xyz[3 * jj + 2] - xyz[3 * ii + 2];
    const float dist = sqrtf(dx * dx + dy * dy + dz * dz);
    const float invd = __builtin_amdgcn_rcpf(fmaxf(dist, 1e-12f));
    float myrbf = 0.0f;
    if (lane < 16) myrbf = __sinf(dist * (float)(lane + 1) * (PI_F / 10.0f)) * invd;
    const float env = (dist < 10.0f) ? 0.5f * (__cosf(PI_F * dist * 0.1f) + 1.0f) : 0.0f;
    float w0 = sb[c0 + 0], w1 = sb[c0 + 1], w2 = sb[c0 + 2], w3 = sb[c0 + 3];
#pragma unroll
    for (int k = 0; k < 16; ++k) {
      const float rk = __shfl(myrbf, k, 32);
      const float4 wr = *(const float4*)&sW[k * FDIM + c0];
      w0 = fmaf(rk, wr.x, w0);
      w1 = fmaf(rk, wr.y, w1);
      w2 = fmaf(rk, wr.z, w2);
      w3 = fmaf(rk, wr.w, w3);
    }
    const float4 pj = *(const float4*)&phi[(size_t)jj * FDIM + c0];
    float* vp = vout + (size_t)ii * FDIM + c0;
    atomicAdd(vp + 0, pj.x * w0 * env);
    atomicAdd(vp + 1, pj.y * w1 * env);
    atomicAdd(vp + 2, pj.z * w2 * env);
    atomicAdd(vp + 3, pj.w * w3 * env);
  }
}

// ---------------------------------------------------------------------------
// Small heads: bb_angle / bb_torsion / sc_angle; builds padded sc_S [N,160]
// ---------------------------------------------------------------------------
__global__ __launch_bounds__(256) void head1_kernel(
    const float* __restrict__ S,
    const float* __restrict__ bbaW1, const float* __restrict__ bbab1,
    const float* __restrict__ bbaW2, const float* __restrict__ bbab2,
    const float* __restrict__ scaW1, const float* __restrict__ scab1,
    const float* __restrict__ scaW2, const float* __restrict__ scab2,
    const float* __restrict__ bbtW1, const float* __restrict__ bbtb1,
    const float* __restrict__ bbtW2, const float* __restrict__ bbtb2,
    float* __restrict__ bbA, float* __restrict__ bbT,
    float* __restrict__ scA, float* __restrict__ scS)
{
  const int n = blockIdx.x * 256 + threadIdx.x;
  if (n >= NN) return;
  const float* s = S + (size_t)n * FDIM;
  float* so = scS + (size_t)n * SCF_PAD;
  float za[3] = { bbab1[0], bbab1[1], bbab1[2] };
  float zt[3] = { bbtb1[0], bbtb1[1], bbtb1[2] };
  float zs[10];
#pragma unroll
  for (int c = 0; c < 10; ++c) zs[c] = scab1[c];
  for (int k = 0; k < FDIM; ++k) {
    const float sv = s[k];
    so[k] = sv;
    const float a = silu_f(sv);
#pragma unroll
    for (int c = 0; c < 3; ++c) {
      za[c] = fmaf(a, bbaW1[k * 3 + c], za[c]);
      zt[c] = fmaf(a, bbtW1[k * 3 + c], zt[c]);
    }
#pragma unroll
    for (int c = 0; c < 10; ++c) zs[c] = fmaf(a, scaW1[k * 10 + c], zs[c]);
  }
  float aa[3], ba[3];
#pragma unroll
  for (int c = 0; c < 3; ++c) aa[c] = silu_f(za[c]);
#pragma unroll
  for (int c = 0; c < 3; ++c) {
    float v = bbab2[c];
#pragma unroll
    for (int r = 0; r < 3; ++r) v = fmaf(aa[r], bbaW2[r * 3 + c], v);
    ba[c] = v;
  }
#pragma unroll
  for (int r = 0; r < 3; ++r) {                 // bbt_in tail = silu(bb_angle)
    const float g = silu_f(ba[r]);
#pragma unroll
    for (int c = 0; c < 3; ++c) zt[c] = fmaf(g, bbtW1[(FDIM + r) * 3 + c], zt[c]);
  }
  float at[3], bt[3];
#pragma unroll
  for (int c = 0; c < 3; ++c) at[c] = silu_f(zt[c]);
#pragma unroll
  for (int c = 0; c < 3; ++c) {
    float v = bbtb2[c];
#pragma unroll
    for (int r = 0; r < 3; ++r) v = fmaf(at[r], bbtW2[r * 3 + c], v);
    bt[c] = v;
  }
  float as[10], sa[10];
#pragma unroll
  for (int c = 0; c < 10; ++c) as[c] = silu_f(zs[c]);
#pragma unroll
  for (int c = 0; c < 10; ++c) {
    float v = scab2[c];
#pragma unroll
    for (int r = 0; r < 10; ++r) v = fmaf(as[r], scaW2[r * 10 + c], v);
    sa[c] = v;
  }
#pragma unroll
  for (int c = 0; c < 3; ++c) { bbA[(size_t)n * 3 + c] = ba[c]; bbT[(size_t)n * 3 + c] = bt[c]; }
#pragma unroll
  for (int c = 0; c < 10; ++c) { scA[(size_t)n * 10 + c] = sa[c]; so[FDIM + c] = sa[c]; }
#pragma unroll
  for (int c = SCF_REAL; c < SCF_PAD; ++c) so[c] = 0.0f;
}

// ft head + final [N,13,3] assembly
__global__ __launch_bounds__(256) void head2_kernel(
    const float* __restrict__ scS, const int* __restrict__ cgz,
    const float* __restrict__ ftW1, const float* __restrict__ ftb1,
    const float* __restrict__ ftW2, const float* __restrict__ ftb2,
    const float* __restrict__ bbDist, const float* __restrict__ scDist,
    const float* __restrict__ bbA, const float* __restrict__ bbT,
    const float* __restrict__ scA, float* __restrict__ out)
{
  const int n = blockIdx.x * 256 + threadIdx.x;
  if (n >= NN) return;
  const float* s = scS + (size_t)n * SCF_PAD;
  float z[10];
#pragma unroll
  for (int c = 0; c < 10; ++c) z[c] = ftb1[c];
  for (int k = 0; k < SCF_REAL; ++k) {
    const float a = silu_f(s[k]);
#pragma unroll
    for (int c = 0; c < 10; ++c) z[c] = fmaf(a, ftW1[k * 10 + c], z[c]);
  }
  float a2[10], st[10];
#pragma unroll
  for (int c = 0; c < 10; ++c) a2[c] = silu_f(z[c]);
#pragma unroll
  for (int c = 0; c < 10; ++c) {
    float v = ftb2[c];
#pragma unroll
    for (int r = 0; r < 10; ++r) v = fmaf(a2[r], ftW2[r * 10 + c], v);
    st[c] = v;
  }
  const int zc = cgz[n];
  float* o = out + (size_t)n * 39;
#pragma unroll
  for (int r = 0; r < 3; ++r) {
    o[r * 3 + 0] = bbDist[(size_t)zc * 3 + r];
    o[r * 3 + 1] = bbA[(size_t)n * 3 + r];
    o[r * 3 + 2] = bbT[(size_t)n * 3 + r];
  }
#pragma unroll
  for (int r = 0; r < 10; ++r) {
    o[9 + r * 3 + 0] = scDist[(size_t)zc * 10 + r];
    o[9 + r * 3 + 1] = scA[(size_t)n * 10 + r];
    o[9 + r * 3 + 2] = st[r];
  }
}

extern "C" void kernel_launch(void* const* d_in, const int* in_sizes, int n_in,
                              void* d_out, int out_size, void* d_ws, size_t ws_size,
                              hipStream_t stream) {
  (void)in_sizes; (void)n_in; (void)out_size; (void)ws_size;
  const int*   cg_z     = (const int*)d_in[0];
  const float* cg_xyz   = (const float*)d_in[1];
  const int*   nbr      = (const int*)d_in[2];
  const float* Sin      = (const float*)d_in[4];
  const float* res_emb  = (const float*)d_in[5];
  const float* msg_W1   = (const float*)d_in[6];
  const float* msg_b1   = (const float*)d_in[7];
  const float* msg_W2   = (const float*)d_in[8];
  const float* msg_b2   = (const float*)d_in[9];
  const float* dist_W   = (const float*)d_in[10];
  const float* dist_b   = (const float*)d_in[11];
  const float* den_W1   = (const float*)d_in[12];
  const float* den_b1   = (const float*)d_in[13];
  const float* den_W2   = (const float*)d_in[14];
  const float* den_b2   = (const float*)d_in[15];
  const float* bb_dist  = (const float*)d_in[16];
  const float* sc_dist  = (const float*)d_in[17];
  const float* bbaW1 = (const float*)d_in[18]; const float* bbab1 = (const float*)d_in[19];
  const float* bbaW2 = (const float*)d_in[20]; const float* bbab2 = (const float*)d_in[21];
  const float* scaW1 = (const float*)d_in[22]; const float* scab1 = (const float*)d_in[23];
  const float* scaW2 = (const float*)d_in[24]; const float* scab2 = (const float*)d_in[25];
  const float* bbtW1 = (const float*)d_in[26]; const float* bbtb1 = (const float*)d_in[27];
  const float* bbtW2 = (const float*)d_in[28]; const float* bbtb2 = (const float*)d_in[29];
  const float* sct_W1 = (const float*)d_in[30]; const float* sct_b1 = (const float*)d_in[31];
  const float* sct_W2 = (const float*)d_in[32]; const float* sct_b2 = (const float*)d_in[33];
  const float* ftW1 = (const float*)d_in[34]; const float* ftb1 = (const float*)d_in[35];
  const float* ftW2 = (const float*)d_in[36]; const float* ftb2 = (const float*)d_in[37];

  // workspace carve (256B aligned)
  char* w = (char*)d_ws;
  auto carve = [&](size_t bytes) { void* p = (void*)w; w += (bytes + 255) & ~(size_t)255; return p; };
  float* Sbuf = (float*)carve((size_t)NN * FDIM * 4);
  float* phi  = (float*)carve((size_t)NN * FDIM * 4);
  float* vbuf = (float*)carve((size_t)NN * FDIM * 4);
  float* scS  = (float*)carve((size_t)NN * SCF_PAD * 4);
  float* bbA  = (float*)carve((size_t)NN * 3 * 4);
  float* bbT  = (float*)carve((size_t)NN * 3 * 4);
  float* scA  = (float*)carve((size_t)NN * 10 * 4);
  const size_t FRAG128 = (size_t)3 * 4 * 8 * 32 * 16;   // halves per buffer (msg/dense)
  const size_t FRAG160 = (size_t)3 * 5 * 9 * 32 * 16;   // halves per buffer (sct)
  unsigned short* msgW1f = (unsigned short*)carve(FRAG128 * 2);
  unsigned short* msgW2f = (unsigned short*)carve(FRAG128 * 2);
  unsigned short* denW1f = (unsigned short*)carve(FRAG128 * 2);
  unsigned short* denW2f = (unsigned short*)carve(FRAG128 * 2);
  unsigned short* sctW1f = (unsigned short*)carve(FRAG160 * 2);
  unsigned short* sctW2f = (unsigned short*)carve(FRAG160 * 2);

  const int gW128 = (3 * 4 * 8 * 32 + 255) / 256;
  const int gW160 = (3 * 5 * 9 * 32 + 255) / 256;
  wconv_kernel<<<gW128, 256, 0, stream>>>(msg_W1, msgW1f, 3, FDIM, FDIM, 4, 8);
  wconv_kernel<<<gW128, 256, 0, stream>>>(msg_W2, msgW2f, 3, FDIM, FDIM, 4, 8);
  wconv_kernel<<<gW128, 256, 0, stream>>>(den_W1, denW1f, 3, FDIM, FDIM, 4, 8);
  wconv_kernel<<<gW128, 256, 0, stream>>>(den_W2, denW2f, 3, FDIM, FDIM, 4, 8);
  wconv_kernel<<<gW160, 256, 0, stream>>>(sct_W1, sctW1f, 3, SCF_REAL, SCF_REAL, 5, 9);
  wconv_kernel<<<gW160, 256, 0, stream>>>(sct_W2, sctW2f, 3, SCF_REAL, SCF_REAL, 5, 9);

  const int gElem = (NN * FDIM + 255) / 256;
  embed_kernel<<<gElem, 256, 0, stream>>>(Sin, cg_z, res_emb, Sbuf);

  const int mlpBlocks = ((NN / 16) + 7) / 8;   // 157
  const size_t lw128 = (size_t)4 * 8 * 32 * 16;
  const size_t lw160 = (size_t)5 * 9 * 32 * 16;

  for (int l = 0; l < 3; ++l) {
    // phi = silu(S @ msg_W1 + b1) @ msg_W2 + b2
    mlp2_kernel<4, 8, 4, 8, FDIM, FDIM, FDIM, FDIM, false, false>
        <<<mlpBlocks, 256, 0, stream>>>(Sbuf, nullptr, phi,
            msgW1f + (size_t)l * lw128, msg_b1 + (size_t)l * FDIM,
            msgW2f + (size_t)l * lw128, msg_b2 + (size_t)l * FDIM, NN);
    zero_kernel<<<gElem, 256, 0, stream>>>(vbuf, NN * FDIM);
    edge_kernel<<<2048, 256, 0, stream>>>(nbr, cg_xyz, phi, vbuf,
        dist_W + (size_t)l * 16 * FDIM, dist_b + (size_t)l * FDIM);
    // S = S + silu(silu(v) @ dense_W1 + b1) @ dense_W2 + b2
    mlp2_kernel<4, 8, 4, 8, FDIM, FDIM, FDIM, FDIM, true, true>
        <<<mlpBlocks, 256, 0, stream>>>(vbuf, Sbuf, Sbuf,
            denW1f + (size_t)l * lw128, den_b1 + (size_t)l * FDIM,
            denW2f + (size_t)l * lw128, den_b2 + (size_t)l * FDIM, NN);
  }

  const int gNode = (NN + 255) / 256;
  head1_kernel<<<gNode, 256, 0, stream>>>(Sbuf,
      bbaW1, bbab1, bbaW2, bbab2, scaW1, scab1, scaW2, scab2,
      bbtW1, bbtb1, bbtW2, bbtb2, bbA, bbT, scA, scS);

  for (int l = 0; l < 3; ++l) {
    // sc_S += silu(silu(sc_S) @ sct_W1 + b1) @ sct_W2 + b2   (padded 160/144)
    mlp2_kernel<5, 9, 5, 9, SCF_PAD, SCF_PAD, SCF_REAL, SCF_REAL, true, true>
        <<<mlpBlocks, 256, 0, stream>>>(scS, scS, scS,
            sctW1f + (size_t)l * lw160, sct_b1 + (size_t)l * SCF_REAL,
            sctW2f + (size_t)l * lw160, sct_b2 + (size_t)l * SCF_REAL, NN);
  }

  head2_kernel<<<gNode, 256, 0, stream>>>(scS, cg_z, ftW1, ftb1, ftW2, ftb2,
      bb_dist, sc_dist, bbA, bbT, scA, (float*)d_out);
}
